// RecursiveEncoder_46153718563045
// MI455X (gfx1250) — compile-verified, matrix operand order checked by reading, not run on hardware
//
#include <hip/hip_runtime.h>

typedef __attribute__((ext_vector_type(16))) _Float16 v16h;
typedef __attribute__((ext_vector_type(8)))  float    v8f;

union AFrag { v16h v; _Float16 h[16]; };
union CFrag { v8f  v; float    f[8];  };

#define USE_ASYNC_STAGE 1

// ---------------------------------------------------------------------------
// WMMA helper (wave32, 16x16x32 f16 -> f32 accumulate)
// ---------------------------------------------------------------------------
__device__ inline v8f wmma_f16(v16h a, v16h b, v8f c) {
  return __builtin_amdgcn_wmma_f32_16x16x32_f16(false, a, false, b,
                                                (short)0, c, false, false);
}

// A fragment (16xK row-major halves, ld in halves). ISA 05_wmma layout:
// lanes 0-15: M=lane, K in {k0..k0+7, k0+16..k0+23}; lanes 16-31: K +8.
__device__ inline v16h load_a_rm(const _Float16* base, int ldh, int k0, int lane) {
  int m  = lane & 15;
  int kb = k0 + ((lane & 16) ? 8 : 0);
  const _Float16* p = base + m * ldh + kb;
  AFrag a;
#pragma unroll
  for (int j = 0; j < 8; ++j) { a.h[j] = p[j]; a.h[j + 8] = p[j + 16]; }
  return a.v;
}

// ---------------------------------------------------------------------------
// Weight pre-pack: [K,N] f32 (optionally two side-by-side N-halves) ->
// f16 fragments: dst[((kt*Ntiles+nt)*32 + lane)*16 + j]  (one 32B load/lane)
// ---------------------------------------------------------------------------
__global__ void k_pack(const float* __restrict__ srcA, const float* __restrict__ srcB,
                       int Ksrc, int NsrcA, int Ktiles, int Ntiles,
                       _Float16* __restrict__ dst) {
  int total = Ktiles * Ntiles * 512;
  for (int e = blockIdx.x * blockDim.x + threadIdx.x; e < total;
       e += gridDim.x * blockDim.x) {
    int j    = e & 15;
    int lane = (e >> 4) & 31;
    int tile = e >> 9;
    int kt = tile / Ntiles, nt = tile % Ntiles;
    int n = nt * 16 + (lane & 15);
    int k = kt * 32 + ((lane & 16) ? 8 : 0) + (j < 8 ? j : 8 + j);
    float v = 0.f;
    if (k < Ksrc) {
      if (srcB != nullptr && n >= NsrcA) v = srcB[k * NsrcA + (n - NsrcA)];
      else                               v = srcA[k * NsrcA + n];
    }
    dst[e] = (_Float16)v;
  }
}

__global__ void k_cvt(const float* __restrict__ src, int n, _Float16* __restrict__ dst) {
  for (int i = blockIdx.x * blockDim.x + threadIdx.x; i < n;
       i += gridDim.x * blockDim.x)
    dst[i] = (_Float16)src[i];
}

// ---------------------------------------------------------------------------
// Fused kernel: per-WG 16 parents.
//   box encoder -> child encoder -> masked max -> W2 -> Ws1 -> [Wmu|Wvar]
//   -> reparameterize + KLD, all on-chip.
// ---------------------------------------------------------------------------
#define K_W1A   0          // 131072 B : packed W1[:256] (staged, reused 10x)
#define K_WBOX  131072     // 16384  B : packed padded W_box
#define K_LEAF  147456     // 8448   B : 16 x 264 halves (leaf tile, then x)
#define K_BOX   155904     // 3200   B : 16 x 100 halves (all box rows, f16)
#define K_SEM   159104     // 640    B
#define K_NCH   159744     // 64     B
#define K_SMEM  159808
// sampler intermediates overlay the dead W1a region:
#define K_P     0          // 8448 B : parent_feat tile (f16, stride 264)
#define K_E     8448       // 8448 B : enc tile
#define K_MV    16896      // 33280 B: 16 x 520 f32 [mu | logvar]

__global__ __launch_bounds__(256)
void k_fused(const float* __restrict__ box, const float* __restrict__ eps,
             const float* __restrict__ b_box, const float* __restrict__ b1,
             const float* __restrict__ b2, const float* __restrict__ bs1,
             const float* __restrict__ bmu, const float* __restrict__ bvar,
             const _Float16* __restrict__ wboxp, const _Float16* __restrict__ w1ap,
             const _Float16* __restrict__ w1s, const _Float16* __restrict__ w2p,
             const _Float16* __restrict__ ws1p, const _Float16* __restrict__ wmvp,
             const int* __restrict__ sem_ids, const int* __restrict__ n_children,
             float* __restrict__ out) {
  extern __shared__ char smem[];
  _Float16* sW1a  = (_Float16*)(smem + K_W1A);
  _Float16* sWbox = (_Float16*)(smem + K_WBOX);
  _Float16* sLeaf = (_Float16*)(smem + K_LEAF);
  _Float16* sBox  = (_Float16*)(smem + K_BOX);
  int*      sSem  = (int*)(smem + K_SEM);
  int*      sNch  = (int*)(smem + K_NCH);
  _Float16* sP  = (_Float16*)(smem + K_P);
  _Float16* sE  = (_Float16*)(smem + K_E);
  float*    sMV = (float*)(smem + K_MV);

  const int t    = threadIdx.x;
  const int lane = t & 31;
  const int wave = t >> 5;
  const int p0   = blockIdx.x * 16;
  const int n_lo = lane & 15;
  const int m_off = (lane & 16) ? 8 : 0;

  // ---- stage packed W1a (8192 x 16B) + Wbox (1024 x 16B) into LDS ----
#if USE_ASYNC_STAGE
  for (int i = t; i < 8192; i += 256) {
    unsigned long long ga = (unsigned long long)(const void*)(w1ap) + (unsigned long long)i * 16ull;
    unsigned la = (unsigned)(K_W1A + i * 16);
    asm volatile("global_load_async_to_lds_b128 %0, %1, off" :: "v"(la), "v"(ga) : "memory");
  }
  for (int i = t; i < 1024; i += 256) {
    unsigned long long ga = (unsigned long long)(const void*)(wboxp) + (unsigned long long)i * 16ull;
    unsigned la = (unsigned)(K_WBOX + i * 16);
    asm volatile("global_load_async_to_lds_b128 %0, %1, off" :: "v"(la), "v"(ga) : "memory");
  }
  asm volatile("s_wait_asynccnt 0" ::: "memory");
#else
  {
    const uint4* gs = (const uint4*)w1ap; uint4* gd = (uint4*)sW1a;
    for (int i = t; i < 8192; i += 256) gd[i] = gs[i];
    gs = (const uint4*)wboxp; gd = (uint4*)sWbox;
    for (int i = t; i < 1024; i += 256) gd[i] = gs[i];
  }
#endif
  // ---- stage all 16 parents' box rows (1600 contiguous f32 -> f16) ----
  {
    const float* gb = box + (size_t)p0 * 100;
    for (int i = t; i < 1600; i += 256) sBox[i] = (_Float16)gb[i];
  }
  if (t < 160) sSem[t] = sem_ids[p0 * 10 + t];
  if (t < 16)  sNch[t] = n_children[p0 + t];
  __syncthreads();

  float xacc[2][8];
#pragma unroll
  for (int q = 0; q < 2; ++q)
#pragma unroll
    for (int r = 0; r < 8; ++r) xacc[q][r] = 0.f;

  // ================= child loop =================
  for (int c = 0; c < 10; ++c) {
    // box A-fragment built directly in registers (K 10 -> 32 zero-padded)
    AFrag abox;
    {
      int m  = lane & 15;
      int kb = (lane & 16) ? 8 : 0;
#pragma unroll
      for (int j = 0; j < 8; ++j) {
        int k = kb + j;
        abox.h[j]     = (k < 10) ? sBox[m * 100 + c * 10 + k] : (_Float16)0.f;
        abox.h[j + 8] = (_Float16)0.f;   // K 16..31 always padded
      }
    }

    // box GEMM (single K step) + bias + relu -> leaf tile in LDS
#pragma unroll
    for (int q = 0; q < 2; ++q) {
      int nt = wave * 2 + q;
      v16h b = *(const v16h*)(sWbox + (nt * 32 + lane) * 16);
      CFrag acc;
#pragma unroll
      for (int r = 0; r < 8; ++r) acc.f[r] = 0.f;
      acc.v = wmma_f16(abox.v, b, acc.v);
      int n = nt * 16 + n_lo;
      float bb = b_box[n];
#pragma unroll
      for (int r = 0; r < 8; ++r) {
        float v = acc.f[r] + bb;
        sLeaf[(r + m_off) * 264 + n] = (_Float16)(v > 0.f ? v : 0.f);
      }
    }
    __syncthreads();

    // child encoder: leaf[16x256] @ W1a + b1 + W1[256+sem] -> relu -> max
#pragma unroll
    for (int q = 0; q < 2; ++q) {
      int nt = wave * 2 + q;
      CFrag acc;
#pragma unroll
      for (int r = 0; r < 8; ++r) acc.f[r] = 0.f;
#pragma unroll
      for (int kt = 0; kt < 8; ++kt) {
        v16h a = load_a_rm(sLeaf, 264, kt * 32, lane);
        v16h b = *(const v16h*)(sW1a + ((kt * 16 + nt) * 32 + lane) * 16);
        acc.v = wmma_f16(a, b, acc.v);
      }
      int n = nt * 16 + n_lo;
      float bb = b1[n];
#pragma unroll
      for (int r = 0; r < 8; ++r) {
        int m = r + m_off;
        int s = sSem[m * 10 + c];
        float v = acc.f[r] + bb + (float)w1s[s * 256 + n];
        v = v > 0.f ? v : 0.f;
        if (c < sNch[m] && v > xacc[q][r]) xacc[q][r] = v;
      }
    }
    __syncthreads();
  }

  // ================= sampler chain (on-chip) =================
  // x = max over children -> reuse leaf tile as the next A operand
#pragma unroll
  for (int q = 0; q < 2; ++q) {
    int n = (wave * 2 + q) * 16 + n_lo;
#pragma unroll
    for (int r = 0; r < 8; ++r)
      sLeaf[(r + m_off) * 264 + n] = (_Float16)xacc[q][r];
  }
  __syncthreads();

  // parent_feat = relu(x @ W2 + b2)
#pragma unroll
  for (int q = 0; q < 2; ++q) {
    int nt = wave * 2 + q;
    CFrag acc;
#pragma unroll
    for (int r = 0; r < 8; ++r) acc.f[r] = 0.f;
#pragma unroll
    for (int kt = 0; kt < 8; ++kt) {
      v16h a = load_a_rm(sLeaf, 264, kt * 32, lane);
      v16h b = *(const v16h*)(w2p + ((kt * 16 + nt) * 32 + lane) * 16);
      acc.v = wmma_f16(a, b, acc.v);
    }
    int n = nt * 16 + n_lo;
    float bb = b2[n];
#pragma unroll
    for (int r = 0; r < 8; ++r) {
      float v = acc.f[r] + bb;
      sP[(r + m_off) * 264 + n] = (_Float16)(v > 0.f ? v : 0.f);
    }
  }
  __syncthreads();

  // enc = relu(parent @ Ws1 + bs1)
#pragma unroll
  for (int q = 0; q < 2; ++q) {
    int nt = wave * 2 + q;
    CFrag acc;
#pragma unroll
    for (int r = 0; r < 8; ++r) acc.f[r] = 0.f;
#pragma unroll
    for (int kt = 0; kt < 8; ++kt) {
      v16h a = load_a_rm(sP, 264, kt * 32, lane);
      v16h b = *(const v16h*)(ws1p + ((kt * 16 + nt) * 32 + lane) * 16);
      acc.v = wmma_f16(a, b, acc.v);
    }
    int n = nt * 16 + n_lo;
    float bb = bs1[n];
#pragma unroll
    for (int r = 0; r < 8; ++r) {
      float v = acc.f[r] + bb;
      sE[(r + m_off) * 264 + n] = (_Float16)(v > 0.f ? v : 0.f);
    }
  }
  __syncthreads();

  // [mu | logvar] = enc @ [Wmu|Wvar]  (N = 512, 4 n-tiles per wave)
#pragma unroll
  for (int q = 0; q < 4; ++q) {
    int nt = wave * 4 + q;
    CFrag acc;
#pragma unroll
    for (int r = 0; r < 8; ++r) acc.f[r] = 0.f;
#pragma unroll
    for (int kt = 0; kt < 8; ++kt) {
      v16h a = load_a_rm(sE, 264, kt * 32, lane);
      v16h b = *(const v16h*)(wmvp + ((kt * 32 + nt) * 32 + lane) * 16);
      acc.v = wmma_f16(a, b, acc.v);
    }
    int n = nt * 16 + n_lo;
    float bb = (n < 256) ? bmu[n] : bvar[n - 256];
#pragma unroll
    for (int r = 0; r < 8; ++r)
      sMV[(r + m_off) * 520 + n] = acc.f[r] + bb;
  }
  __syncthreads();

  // epilogue: out[:, :256] = eps*exp(0.5*lv)+mu ; out[:, 256:] = 1+lv-mu^2-exp(lv)
  for (int e = t; e < 4096; e += 256) {
    int m = e >> 8, n = e & 255;
    float mu = sMV[m * 520 + n];
    float lv = sMV[m * 520 + 256 + n];
    float ep = eps[(size_t)(p0 + m) * 256 + n];
    float st = expf(0.5f * lv);
    float el = expf(lv);
    size_t ro = (size_t)(p0 + m) * 512;
    out[ro + n]       = ep * st + mu;
    out[ro + 256 + n] = 1.f + lv - mu * mu - el;
  }
}

// ---------------------------------------------------------------------------
// Host-side orchestration
// ---------------------------------------------------------------------------
extern "C" void kernel_launch(void* const* d_in, const int* in_sizes, int n_in,
                              void* d_out, int out_size, void* d_ws, size_t ws_size,
                              hipStream_t stream) {
  const float* box  = (const float*)d_in[0];
  const float* eps  = (const float*)d_in[1];
  const float* Wbox = (const float*)d_in[2];
  const float* bbox = (const float*)d_in[3];
  const float* W1   = (const float*)d_in[4];
  const float* b1   = (const float*)d_in[5];
  const float* W2   = (const float*)d_in[6];
  const float* b2   = (const float*)d_in[7];
  const float* Ws1  = (const float*)d_in[8];
  const float* bs1  = (const float*)d_in[9];
  const float* Wmu  = (const float*)d_in[10];
  const float* bmu  = (const float*)d_in[11];
  const float* Wvar = (const float*)d_in[12];
  const float* bvar = (const float*)d_in[13];
  const int*   sem  = (const int*)d_in[14];
  const int*   nch  = (const int*)d_in[15];
  float* out = (float*)d_out;

  char* ws = (char*)d_ws;
  _Float16* wboxp = (_Float16*)(ws);                     // 16 KB
  _Float16* w1ap  = (_Float16*)(ws + 16384);             // 128 KB
  _Float16* w1s   = (_Float16*)(ws + 147456);            // 32 KB
  _Float16* w2p   = (_Float16*)(ws + 180224);            // 128 KB
  _Float16* ws1p  = (_Float16*)(ws + 311296);            // 128 KB
  _Float16* wmvp  = (_Float16*)(ws + 442368);            // 256 KB

  const int B = in_sizes[15];          // n_children: one entry per parent
  const int blocks = B / 16;

  // weight re-pack into WMMA fragment layout (tiny, L2-resident)
  k_pack<<<32,  256, 0, stream>>>(Wbox, nullptr, 10, 256, 1, 16, wboxp);
  k_pack<<<256, 256, 0, stream>>>(W1,   nullptr, 256, 256, 8, 16, w1ap);
  k_cvt <<<64,  256, 0, stream>>>(W1 + 256 * 256, 64 * 256, w1s);
  k_pack<<<256, 256, 0, stream>>>(W2,   nullptr, 256, 256, 8, 16, w2p);
  k_pack<<<256, 256, 0, stream>>>(Ws1,  nullptr, 256, 256, 8, 16, ws1p);
  k_pack<<<512, 256, 0, stream>>>(Wmu,  Wvar,    256, 256, 8, 32, wmvp);

  k_fused<<<blocks, 256, K_SMEM, stream>>>(box, eps, bbox, b1, b2, bs1, bmu, bvar,
                                           wboxp, w1ap, w1s, w2p, ws1p, wmvp,
                                           sem, nch, out);
}
